// GAT_61649960567469
// MI455X (gfx1250) — compile-verified, matrix operand order checked by reading, not run on hardware
//
#include <hip/hip_runtime.h>
#include <hip/hip_bf16.h>

#define NN 50000
#define NE 800000
#define INDIM 128
#define HID 32
#define HEADS 8
#define OUTD 64
#define NEG_SLOPE 0.2f
#define EPSV 1e-16f

typedef __attribute__((ext_vector_type(16))) _Float16 v16h;
typedef __attribute__((ext_vector_type(8)))  float    v8f;

// ---- order-preserving float<->uint encoding (for atomicMax on floats) ----
__device__ __forceinline__ unsigned f2ord(float f) {
    unsigned u = __float_as_uint(f);
    return (u & 0x80000000u) ? ~u : (u | 0x80000000u);
}
__device__ __forceinline__ float ord2f(unsigned u) {
    return __uint_as_float((u & 0x80000000u) ? (u & 0x7FFFFFFFu) : ~u);
}

// ---- generic fills (accumulators must be re-initialized every call) ----
__global__ void fill_f32(float* p, float v, int n) {
    int i = blockIdx.x * blockDim.x + threadIdx.x;
    if (i < n) p[i] = v;
}
__global__ void fill_u32(unsigned* p, unsigned v, int n) {
    int i = blockIdx.x * blockDim.x + threadIdx.x;
    if (i < n) p[i] = v;
}

// ---- load a 16-elem K-slice (WMMA 16-bit fragment layout) and split f32 -> hi+lo f16 ----
// 16-bit A/B layout per lane: half 0 holds K = {0..7, 16..23}, half 1 holds K = {8..15, 24..31}
__device__ __forceinline__ void load_split(const float* __restrict__ p, int k0, int kb,
                                           v16h& hi, v16h& lo) {
#pragma unroll
    for (int j = 0; j < 8; ++j) {
        float v0 = p[k0 + kb + j];
        float v1 = p[k0 + 16 + kb + j];
        _Float16 h0 = (_Float16)v0;
        _Float16 h1 = (_Float16)v1;
        hi[j]     = h0;
        hi[8 + j] = h1;
        lo[j]     = (_Float16)(v0 - (float)h0);
        lo[8 + j] = (_Float16)(v1 - (float)h1);
    }
}

// ---- C[N x outDim] = A[N x K] @ W[outDim x K]^T ----
// Error-compensated f16 WMMA: A = ah+al, B = bh+bl (f16 splits);
// acc += ah*bh + al*bh + ah*bl  (drops only the ~2^-22 al*bl term -> near-f32 accuracy).
// One wave computes a 16x64 tile (4 accumulators; A fragment loaded once, reused 4x).
// block = 128 (4 waves, consecutive row tiles); grid.y covers outDim in 64-col groups.
__global__ void gemm_nt_wmma(const float* __restrict__ A, const float* __restrict__ W,
                             float* __restrict__ C, int K, int outDim, int numRT) {
    const int wave = threadIdx.x >> 5;
    const int lane = threadIdx.x & 31;
    const int half = lane >> 4;
    const int r    = lane & 15;
    const int rt   = blockIdx.x * 4 + wave;
    if (rt >= numRT) return;
    const int rowBase = rt * 16;
    const int colBase = blockIdx.y * 64;
    const int kb = half * 8;

    const float* arow = A + (size_t)(rowBase + r) * K;

    v8f acc[4] = {{}, {}, {}, {}};
    for (int k0 = 0; k0 < K; k0 += 32) {
        v16h ah, al;
        load_split(arow, k0, kb, ah, al);
#pragma unroll
        for (int c = 0; c < 4; ++c) {
            const float* brow = W + (size_t)(colBase + c * 16 + r) * K;  // B[k][n] = W[n][k]
            v16h bh, bl;
            load_split(brow, k0, kb, bh, bl);
            acc[c] = __builtin_amdgcn_wmma_f32_16x16x32_f16(false, ah, false, bh,
                                                            (short)0, acc[c], false, false);
            acc[c] = __builtin_amdgcn_wmma_f32_16x16x32_f16(false, al, false, bh,
                                                            (short)0, acc[c], false, false);
            acc[c] = __builtin_amdgcn_wmma_f32_16x16x32_f16(false, ah, false, bl,
                                                            (short)0, acc[c], false, false);
        }
    }
    // C/D layout: VGPR j -> row j + 8*half, lane r -> col
#pragma unroll
    for (int c = 0; c < 4; ++c)
#pragma unroll
        for (int j = 0; j < 8; ++j)
            C[(size_t)(rowBase + j + half * 8) * outDim + colBase + c * 16 + r] = acc[c][j];
}

// ---- a_src[n,h] = <h[n,h,:], att_src[h,:]>, same for dst ----
__global__ void att_dots(const float* __restrict__ h, const float* __restrict__ att_src,
                         const float* __restrict__ att_dst, float* __restrict__ asrc,
                         float* __restrict__ adst, int heads, int dim) {
    int idx = blockIdx.x * blockDim.x + threadIdx.x;
    if (idx >= NN * heads) return;
    int n = idx / heads, hd = idx % heads;
    const float* hp = h + (size_t)n * heads * dim + (size_t)hd * dim;
    const float* as = att_src + hd * dim;
    const float* ad = att_dst + hd * dim;
    float s0 = 0.f, s1 = 0.f;
    for (int d = 0; d < dim; ++d) { float v = hp[d]; s0 += v * as[d]; s1 += v * ad[d]; }
    asrc[idx] = s0; adst[idx] = s1;
}

// ---- pass 1: segment max of leaky_relu logits over dst (incl. self loops) ----
__global__ void edge_max(const int* __restrict__ ei, const float* __restrict__ asrc,
                         const float* __restrict__ adst, unsigned* __restrict__ m, int heads) {
    long long idx = (long long)blockIdx.x * blockDim.x + threadIdx.x;
    long long tot = (long long)(NE + NN) * heads;
    if (idx >= tot) return;
    int e = (int)(idx / heads), hd = (int)(idx % heads);
    int src = (e < NE) ? ei[e]      : e - NE;
    int dst = (e < NE) ? ei[NE + e] : e - NE;
    float lg = asrc[src * heads + hd] + adst[dst * heads + hd];
    lg = lg > 0.f ? lg : NEG_SLOPE * lg;
    atomicMax(&m[dst * heads + hd], f2ord(lg));
}

// ---- pass 2 (fused): e = exp(logit - m); s[dst] += e; acc[dst] += e * h[src] ----
// one block of (heads*dim) threads per edge; prefetch the gather row so its L2
// latency overlaps the logit/exp computation.
__global__ void edge_acc(const int* __restrict__ ei, const float* __restrict__ asrc,
                         const float* __restrict__ adst, const unsigned* __restrict__ m,
                         const float* __restrict__ h, float* __restrict__ s,
                         float* __restrict__ acc, int heads, int dim) {
    int e = blockIdx.x;
    int t = threadIdx.x;          // t = hd*dim + d
    int hd = t / dim, d = t % dim;
    int src = (e < NE) ? ei[e]      : e - NE;
    int dst = (e < NE) ? ei[NE + e] : e - NE;
    const float* hp = h + (size_t)src * heads * dim + t;
    __builtin_prefetch(hp, 0, 0);                       // global_prefetch_b8
    float lg = asrc[src * heads + hd] + adst[dst * heads + hd];
    lg = lg > 0.f ? lg : NEG_SLOPE * lg;
    float ex = __expf(lg - ord2f(m[dst * heads + hd]));
    if (d == 0) atomicAdd(&s[dst * heads + hd], ex);
    atomicAdd(&acc[(size_t)dst * heads * dim + t], ex * (*hp));
}

// ---- layer-1 finalize (in place): h2 = relu(acc / (s+eps) + b1) ----
__global__ void finalize1(float* __restrict__ acc, const float* __restrict__ s,
                          const float* __restrict__ b) {
    int i = blockIdx.x * blockDim.x + threadIdx.x;
    if (i >= NN * HEADS * HID) return;
    int n = i / (HEADS * HID);
    int c = i % (HEADS * HID);
    int hd = c / HID;
    float v = acc[i] / (s[n * HEADS + hd] + EPSV) + b[c];
    acc[i] = v > 0.f ? v : 0.f;
}

// ---- layer-2 finalize: out = acc / (s+eps) + b2  (mean over 1 head == identity) ----
__global__ void finalize2(float* __restrict__ out, const float* __restrict__ s,
                          const float* __restrict__ b) {
    int i = blockIdx.x * blockDim.x + threadIdx.x;
    if (i >= NN * OUTD) return;
    int n = i / OUTD, d = i % OUTD;
    out[i] = out[i] / (s[n] + EPSV) + b[d];
}

static inline int cdiv(long long a, int b) { return (int)((a + b - 1) / b); }

extern "C" void kernel_launch(void* const* d_in, const int* in_sizes, int n_in,
                              void* d_out, int out_size, void* d_ws, size_t ws_size,
                              hipStream_t stream) {
    (void)in_sizes; (void)n_in; (void)out_size; (void)ws_size;
    const float* x        = (const float*)d_in[0];
    const int*   ei       = (const int*)d_in[1];   // [2,E] row-major
    const float* W1       = (const float*)d_in[2];
    const float* att_src1 = (const float*)d_in[3];
    const float* att_dst1 = (const float*)d_in[4];
    const float* b1       = (const float*)d_in[5];
    const float* W2       = (const float*)d_in[6];
    const float* att_src2 = (const float*)d_in[7];
    const float* att_dst2 = (const float*)d_in[8];
    const float* b2       = (const float*)d_in[9];
    float* out = (float*)d_out;

    // workspace carve-up (~110 MB)
    float*    h1    = (float*)d_ws;                 // [N,256]; later reused as h3 [N,64]
    float*    acc1  = h1   + (size_t)NN * 256;      // [N,256] accumulator -> h2 in place
    float*    asrc1 = acc1 + (size_t)NN * 256;      // [N,8]
    float*    adst1 = asrc1 + (size_t)NN * 8;       // [N,8]
    unsigned* m1    = (unsigned*)(adst1 + (size_t)NN * 8);  // [N,8]
    float*    s1    = (float*)(m1 + (size_t)NN * 8);        // [N,8]
    float*    asrc2 = s1 + (size_t)NN * 8;          // [N]
    float*    adst2 = asrc2 + NN;                   // [N]
    unsigned* m2    = (unsigned*)(adst2 + NN);      // [N]
    float*    s2    = (float*)(m2 + NN);            // [N]
    float*    h3    = h1;                           // [N,64] alias (h1 dead by then)

    const int EDG   = NE + NN;
    const int numRT = NN / 16;                      // 3125 row tiles (exact)

    // ---------------- layer 1 ----------------
    gemm_nt_wmma<<<dim3(cdiv(numRT, 4), 256 / 64), 128, 0, stream>>>(x, W1, h1, INDIM, 256, numRT);
    att_dots<<<cdiv((long long)NN * HEADS, 256), 256, 0, stream>>>(h1, att_src1, att_dst1,
                                                                   asrc1, adst1, HEADS, HID);
    fill_u32<<<cdiv(NN * HEADS, 256), 256, 0, stream>>>(m1, 0u, NN * HEADS);   // 0 == encode(<=-inf)
    fill_f32<<<cdiv(NN * HEADS, 256), 256, 0, stream>>>(s1, 0.f, NN * HEADS);
    fill_f32<<<cdiv(NN * 256, 256), 256, 0, stream>>>(acc1, 0.f, NN * 256);
    edge_max<<<cdiv((long long)EDG * HEADS, 256), 256, 0, stream>>>(ei, asrc1, adst1, m1, HEADS);
    edge_acc<<<EDG, HEADS * HID, 0, stream>>>(ei, asrc1, adst1, m1, h1, s1, acc1, HEADS, HID);
    finalize1<<<cdiv(NN * 256, 256), 256, 0, stream>>>(acc1, s1, b1);          // acc1 is now h2

    // ---------------- layer 2 ----------------
    gemm_nt_wmma<<<dim3(cdiv(numRT, 4), 1), 128, 0, stream>>>(acc1, W2, h3, 256, OUTD, numRT);
    att_dots<<<cdiv(NN, 256), 256, 0, stream>>>(h3, att_src2, att_dst2, asrc2, adst2, 1, OUTD);
    fill_u32<<<cdiv(NN, 256), 256, 0, stream>>>(m2, 0u, NN);
    fill_f32<<<cdiv(NN, 256), 256, 0, stream>>>(s2, 0.f, NN);
    fill_f32<<<cdiv(NN * OUTD, 256), 256, 0, stream>>>(out, 0.f, NN * OUTD);
    edge_max<<<cdiv(EDG, 256), 256, 0, stream>>>(ei, asrc2, adst2, m2, 1);
    edge_acc<<<EDG, OUTD, 0, stream>>>(ei, asrc2, adst2, m2, h3, s2, out, 1, OUTD);
    finalize2<<<cdiv(NN * OUTD, 256), 256, 0, stream>>>(out, s2, b2);
}
